// GraphTransformer_10402410790973
// MI455X (gfx1250) — matrix operand order repaired and checked by reading, not compiled, not run on hardware
//
#include <hip/hip_runtime.h>

// ---------------------------------------------------------------------------
// Graph Transformer forward for MI455X (gfx1250, wave32, WMMA + TDM).
// B=1, N=384, DIM=256, HEADS=8, DHEAD=64, INNER=512, FF=1024, EDGE_DIM=64.
//
// Algebraic refactor (exact up to bf16 rounding of WMMA operands):
//   sim[h,i,j] = SCALE*( q_i·k_j + edges_n[i,j]·(W_ekv_h^T q_i) )   [q·b_ekv is
//                softmax-invariant and dropped]
//   out[h,i]   = attn_i@V + (attn_i@E_i)@W_ekv_h + b_ekv_h
// so the 302MB e_kv tensor is never materialized.
// ---------------------------------------------------------------------------

#define NTOK   384
#define DIMN   256
#define HEADS  8
#define DHEAD  64
#define INNERD 512
#define FFD    1024
#define EDIM   64
#define NN     (NTOK * NTOK)
#define SCL    0.125f        // 64^-0.5
#define LN_EPS 1e-5f

typedef __attribute__((ext_vector_type(16))) __bf16 v16bf;
typedef __attribute__((ext_vector_type(8)))  float  v8f;
typedef unsigned int __attribute__((ext_vector_type(4))) u32x4;
typedef int __attribute__((ext_vector_type(4))) i32x4;
typedef int __attribute__((ext_vector_type(8))) i32x8;

#ifndef __has_builtin
#define __has_builtin(x) 0
#endif
#if __has_builtin(__builtin_amdgcn_tensor_load_to_lds) && \
    __has_builtin(__builtin_amdgcn_s_wait_tensorcnt)
#define HAVE_TDM 1
#if defined(__has_include)
#if __has_include(<hip/amd_detail/amd_gfx1250_TDM.h>)
#define TDM_6ARG 1
#endif
#endif
#endif

__device__ __forceinline__ __bf16 f2bf(float f) { return (__bf16)f; }

__device__ __forceinline__ float wave_sum(float v) {
#pragma unroll
  for (int o = 16; o > 0; o >>= 1) v += __shfl_xor(v, o, 32);
  return v;
}
__device__ __forceinline__ float wave_max(float v) {
#pragma unroll
  for (int o = 16; o > 0; o >>= 1) v = fmaxf(v, __shfl_xor(v, o, 32));
  return v;
}

// ---------------------------------------------------------------------------
// Generic fully-strided batched GEMM on v_wmma_f32_16x16x32_bf16.
//   C[m*csm + n*csn + b*cb] (+)= act( sum_k A[m*am + k*ak + b*ab] *
//                                           W[k*wk + n*wn + b*wb] + bias[n] )
// One wave computes a 16x64 strip: the A fragment is loaded once per K-chunk
// and reused by 4 WMMAs (4 accumulators). The 32x64 B tile is staged through
// LDS with coalesced float4 row loads (pitch 69 -> conflict-free fragment
// reads for both lane halves). Ragged tiles (M=8 / Nc=8 launches) use a
// clamped-address path masked with v_cndmask (no exec-branchy loads).
// Fragment layouts per CDNA5 ISA 7.12.2 (wave32):
//   A 16x32 bf16 : row = lane&15 ; elems 0..7 -> K = hi*8+e, 8..15 -> 16+hi*8+e
//   B 32x16 bf16 : col = lane&15 ; elems 0..7 -> K = hi*8+e, 8..15 -> 16+hi*8+e
//   C/D 16x16 f32: row = hi*8+g  ; col = lane&15
// ---------------------------------------------------------------------------
__global__ __launch_bounds__(32)
void gemm_wmma_bf16(const float* __restrict__ A, int am, int ak, long long ab,
                    const float* __restrict__ W, int wk, int wn, long long wb,
                    const float* __restrict__ bias, int bn, long long bb,
                    float* __restrict__ C, int csm, int csn, long long cb,
                    int M, int Nc, int K, int act, int accum)
{
  __shared__ float Bs[32][69];

  const int lane = threadIdx.x;
  const int r    = lane & 15;
  const int hi   = lane >> 4;
  const int tm   = blockIdx.y * 16;
  const int tn0  = blockIdx.x * 64;
  const long long b = blockIdx.z;

  A += b * ab;
  W += b * wb;
  C += b * cb;
  if (bias) bias += b * bb;

  v8f acc[4] = {{}, {}, {}, {}};

  if ((tm + 16 <= M) && (tn0 + 64 <= Nc)) {
    // ---------------- fast path: full 16x64 strip ----------------
    const float* Arow = A + (long long)(tm + r) * am;
    for (int k0 = 0; k0 < K; k0 += 32) {
      // stage B 32x64 tile into LDS (coalesced row-major float4 loads)
      if (wn == 1) {
        const float* Wb = W + (long long)k0 * wk + tn0;
#pragma unroll
        for (int t = 0; t < 16; ++t) {
          const int idx = t * 32 + lane;          // 0..511 float4 slots
          const int row = idx >> 4;               // 0..31
          const int c4  = (idx & 15) * 4;         // 0..60
          const float4 wv = *(const float4*)(Wb + (long long)row * wk + c4);
          Bs[row][c4 + 0] = wv.x; Bs[row][c4 + 1] = wv.y;
          Bs[row][c4 + 2] = wv.z; Bs[row][c4 + 3] = wv.w;
        }
      } else {
#pragma unroll
        for (int t = 0; t < 16; ++t) {
          const int idx = t * 32 + lane;
          const int row = idx >> 4;
          const int c4  = (idx & 15) * 4;
          const float* wp = W + (long long)(k0 + row) * wk + (long long)(tn0 + c4) * wn;
          Bs[row][c4 + 0] = wp[0];      Bs[row][c4 + 1] = wp[wn];
          Bs[row][c4 + 2] = wp[2 * wn]; Bs[row][c4 + 3] = wp[3 * wn];
        }
      }
      if (k0 + 32 < K)                  // stream hint for next B chunk
        __builtin_prefetch((const void*)&W[(long long)(k0 + 32) * wk +
                                           (long long)tn0 * wn + lane], 0, 3);
      __syncthreads();                  // single-wave WG: S_NOP + fence

      v16bf af;                         // one A fragment feeds 4 WMMAs
      if (ak == 1) {
        const float4 a0 = *(const float4*)(Arow + k0 + hi * 8);
        const float4 a1 = *(const float4*)(Arow + k0 + hi * 8 + 4);
        const float4 a2 = *(const float4*)(Arow + k0 + 16 + hi * 8);
        const float4 a3 = *(const float4*)(Arow + k0 + 16 + hi * 8 + 4);
        af[0] = f2bf(a0.x);  af[1] = f2bf(a0.y);  af[2]  = f2bf(a0.z);  af[3]  = f2bf(a0.w);
        af[4] = f2bf(a1.x);  af[5] = f2bf(a1.y);  af[6]  = f2bf(a1.z);  af[7]  = f2bf(a1.w);
        af[8] = f2bf(a2.x);  af[9] = f2bf(a2.y);  af[10] = f2bf(a2.z);  af[11] = f2bf(a2.w);
        af[12] = f2bf(a3.x); af[13] = f2bf(a3.y); af[14] = f2bf(a3.z);  af[15] = f2bf(a3.w);
      } else {
#pragma unroll
        for (int e = 0; e < 8; ++e) {
          af[e]     = f2bf(Arow[(long long)(k0 + hi * 8 + e) * ak]);
          af[e + 8] = f2bf(Arow[(long long)(k0 + 16 + hi * 8 + e) * ak]);
        }
      }
#pragma unroll
      for (int s = 0; s < 4; ++s) {
        v16bf bf;
#pragma unroll
        for (int e = 0; e < 16; ++e) bf[e] = f2bf(Bs[hi * 8 + e + (e >= 8 ? 8 : 0)][s * 16 + r]);
        acc[s] = __builtin_amdgcn_wmma_f32_16x16x32_bf16(false, af, false, bf,
                                                         (short)0, acc[s], false, false);
      }
      __syncthreads();
    }
  } else {
    // ---------------- ragged path: clamped addresses + select masks --------
    const int  mrow = tm + r;
    const bool mv = mrow < M;
    const long long mrc = (long long)(mv ? mrow : 0) * am;
#pragma unroll
    for (int s = 0; s < 4; ++s) {
      const int tn = tn0 + s * 16;
      if (tn >= Nc) continue;
      const int  ncol = tn + r;
      const bool nv = ncol < Nc;
      const long long ncc = (long long)(nv ? ncol : 0) * wn;
      for (int k0 = 0; k0 < K; k0 += 32) {
        v16bf af, bf;
#pragma unroll
        for (int e = 0; e < 8; ++e) {
          const float t0 = A[mrc + (long long)(k0 + hi * 8 + e) * ak];
          const float t1 = A[mrc + (long long)(k0 + 16 + hi * 8 + e) * ak];
          af[e]     = f2bf(mv ? t0 : 0.f);
          af[e + 8] = f2bf(mv ? t1 : 0.f);
        }
#pragma unroll
        for (int e = 0; e < 16; ++e) {
          const float t = W[(long long)(k0 + hi * 8 + e + (e >= 8 ? 8 : 0)) * wk + ncc];
          bf[e] = f2bf(nv ? t : 0.f);
        }
        acc[s] = __builtin_amdgcn_wmma_f32_16x16x32_bf16(false, af, false, bf,
                                                         (short)0, acc[s], false, false);
      }
    }
  }

  // ---------------- epilogue: bias / GELU / accumulate / store -------------
#pragma unroll
  for (int s = 0; s < 4; ++s) {
    const int ncol = tn0 + s * 16 + r;
    if (ncol >= Nc) continue;
    const float bv = bias ? bias[(long long)ncol * bn] : 0.f;
#pragma unroll
    for (int g = 0; g < 8; ++g) {
      const int cm = tm + hi * 8 + g;
      if (cm < M) {
        float v = acc[s][g] + bv;
        if (act == 1)                    // exact GELU: 0.5*x*(1+erf(x/sqrt(2)))
          v = 0.5f * v * (1.0f + erff(v * 0.70710678118654752f));
        const long long ci = (long long)cm * csm + (long long)ncol * csn;
        if (accum) v += C[ci];
        C[ci] = v;
      }
    }
  }
}

// --------------------------- LayerNorm (rows) ------------------------------
__global__ __launch_bounds__(32)
void ln_kernel(const float* __restrict__ x, const float* __restrict__ g,
               const float* __restrict__ b, float* __restrict__ y, int D)
{
  const int n = blockIdx.x, lane = threadIdx.x;
  const float* xr = x + (long long)n * D;
  float s = 0.f;
  for (int d = lane; d < D; d += 32) s += xr[d];
  const float mean = wave_sum(s) / (float)D;
  float v = 0.f;
  for (int d = lane; d < D; d += 32) { float t = xr[d] - mean; v += t * t; }
  const float inv = rsqrtf(wave_sum(v) / (float)D + LN_EPS);
  float* yr = y + (long long)n * D;
  for (int d = lane; d < D; d += 32) yr[d] = (xr[d] - mean) * inv * g[d] + b[d];
}

// ------------------- edge LayerNorm over channel dim -----------------------
__global__ __launch_bounds__(256)
void edge_ln_kernel(const float* __restrict__ e, const float* __restrict__ g,
                    const float* __restrict__ b, float* __restrict__ out_cij,
                    float* __restrict__ out_ijc)
{
  const int ij = blockIdx.x * 256 + threadIdx.x;
  if (ij >= NN) return;
  float vals[EDIM];
  float s = 0.f;
#pragma unroll
  for (int c = 0; c < EDIM; ++c) { vals[c] = e[(long long)c * NN + ij]; s += vals[c]; }
  const float mean = s * (1.f / EDIM);
  float v = 0.f;
#pragma unroll
  for (int c = 0; c < EDIM; ++c) { float t = vals[c] - mean; v += t * t; }
  const float inv = rsqrtf(v * (1.f / EDIM) + LN_EPS);
#pragma unroll
  for (int c = 0; c < EDIM; ++c) {
    const float r = (vals[c] - mean) * inv * g[c] + b[c];
    out_cij[(long long)c * NN + ij] = r;
    out_ijc[(long long)ij * EDIM + c] = r;
  }
}

// --------------------------- RoPE on q and k -------------------------------
__global__ __launch_bounds__(256)
void rope_kernel(float* __restrict__ q, float* __restrict__ kv)
{
  const int idx = blockIdx.x * 256 + threadIdx.x;   // pair index
  if (idx >= NTOK * (INNERD / 2)) return;
  const int n = idx / (INNERD / 2);
  const int p = idx % (INNERD / 2);                 // h*32 + m
  const int h = p >> 5, m = p & 31;
  const float f = (float)n * powf(10000.f, -(float)(2 * m) / 64.f);
  const float c = cosf(f), s = sinf(f);
  const int d0 = h * DHEAD + 2 * m;
  const long long qi = (long long)n * INNERD + d0;
  const float q0 = q[qi], q1 = q[qi + 1];
  q[qi]     = q0 * c - q1 * s;
  q[qi + 1] = q1 * c + q0 * s;
  const long long ki = (long long)n * (2 * INNERD) + d0;
  const float k0 = kv[ki], k1 = kv[ki + 1];
  kv[ki]     = k0 * c - k1 * s;
  kv[ki + 1] = k1 * c + k0 * s;
}

// --------------------- fused attention core (per head) ---------------------
// grid: (N/16, HEADS), 1 wave/block.
// The 16x64 q tile is staged to LDS by the Tensor Data Mover (2-D descriptor,
// data_size=4B, pad_enable: 1 dword of LDS padding per 64-dword row so the
// staged tile lands with a 65-float pitch -> conflict-free fragment reads).
// q fragments are then hoisted to registers (invariant over all 24 j-tiles).
// K tiles / V chunks are staged through LDS with coalesced float4 loads.
__global__ __launch_bounds__(32)
void attn_kernel(const float* __restrict__ q, const float* __restrict__ kv,
                 const float* __restrict__ SE, float* __restrict__ attn,
                 float* __restrict__ aout)
{
  __shared__ float Qs[16][DHEAD + 1];   // TDM-padded pitch
  __shared__ float KVs[32][DHEAD + 1];  // K tile (16 rows) / V chunk (32 rows)
  __shared__ float S[16][NTOK + 4];
  const int lane = threadIdx.x;
  const int r = lane & 15, hi = lane >> 4;
  const int h  = blockIdx.y;
  const int i0 = blockIdx.x * 16;

  // ---- stage q tile (16 rows x 64 f32) into LDS via TDM -------------------
#if defined(HAVE_TDM)
  {
    const unsigned lds_off = (unsigned)(unsigned long long)&Qs[0][0];
    const unsigned long long ga =
        (unsigned long long)(const void*)(q + (long long)i0 * INNERD + h * DHEAD);
    u32x4 g0 = {};
    g0[0] = 1u;                                   // count=1, user descriptor
    g0[1] = lds_off;                              // lds_addr
    g0[2] = (unsigned)(ga & 0xffffffffull);       // global_addr[31:0]
    g0[3] = (unsigned)((ga >> 32) & 0x01ffffffull) | (2u << 30); // [56:32]|type=2
    i32x8 g1 = {};
    g1[0] = (int)((2u << 16) | (1u << 20) | (5u << 22)); // 4B elems, pad 1dw/64dw
    g1[1] = (int)((unsigned)INNERD << 16);        // tensor_dim0 = 512 (lo16)
    g1[2] = (int)((unsigned)NTOK << 16);          // dim0 hi=0 | tensor_dim1=384
    g1[3] = (int)(64u << 16);                     // dim1 hi=0 | tile_dim0 = 64
    g1[4] = 16;                                   // tile_dim1 = 16 (tile_dim2=0)
    g1[5] = INNERD;                               // tensor_dim0_stride = 512
    i32x4 gz = {};
#if defined(TDM_6ARG)
    i32x8 gz8 = {};
    __builtin_amdgcn_tensor_load_to_lds(g0, g1, gz, gz, gz8, 0);
#else
    __builtin_amdgcn_tensor_load_to_lds(g0, g1, gz, gz, 0);
#endif
    __builtin_amdgcn_s_wait_tensorcnt((short)0);
  }
#else
  for (int t = lane; t < 16 * (DHEAD / 4); t += 32) {
    const int row = t >> 4, c4 = (t & 15) * 4;
    const float4 v = *(const float4*)(q + (long long)(i0 + row) * INNERD +
                                      h * DHEAD + c4);
    Qs[row][c4 + 0] = v.x; Qs[row][c4 + 1] = v.y;
    Qs[row][c4 + 2] = v.z; Qs[row][c4 + 3] = v.w;
  }
#endif
  __syncthreads();

  // hoist the two q K-chunk fragments (reused by all 24 j-tiles)
  v16bf qf[2];
#pragma unroll
  for (int kc = 0; kc < 2; ++kc) {
#pragma unroll
    for (int e = 0; e < 8; ++e) {
      qf[kc][e]     = f2bf(Qs[r][kc * 32 + hi * 8 + e]);
      qf[kc][e + 8] = f2bf(Qs[r][kc * 32 + 16 + hi * 8 + e]);
    }
  }

  // Phase 1: S[i,j] = SCL * (q_i . k_j + SE[h,i,j])
  const float* SErow = SE + (long long)h * NN + (long long)i0 * NTOK;
  for (int jt = 0; jt < NTOK / 16; ++jt) {
    // stage K rows jt*16..+15 (coalesced)
#pragma unroll
    for (int t = 0; t < 8; ++t) {
      const int idx = t * 32 + lane;            // 0..255 float4 slots
      const int row = idx >> 4, c4 = (idx & 15) * 4;
      const float4 v = *(const float4*)(kv + (long long)(jt * 16 + row) * (2 * INNERD) +
                                        h * DHEAD + c4);
      KVs[row][c4 + 0] = v.x; KVs[row][c4 + 1] = v.y;
      KVs[row][c4 + 2] = v.z; KVs[row][c4 + 3] = v.w;
    }
    __syncthreads();
    v8f acc = {};
#pragma unroll
    for (int kc = 0; kc < 2; ++kc) {
      v16bf bf;
#pragma unroll
      for (int e = 0; e < 16; ++e) bf[e] = f2bf(KVs[r][kc * 32 + hi * 8 + e + (e >= 8 ? 8 : 0)]);
      acc = __builtin_amdgcn_wmma_f32_16x16x32_bf16(false, qf[kc], false, bf,
                                                    (short)0, acc, false, false);
    }
#pragma unroll
    for (int g = 0; g < 8; ++g) {
      const int row = hi * 8 + g;
      S[row][jt * 16 + r] = (acc[g] + SErow[(long long)row * NTOK + jt * 16 + r]) * SCL;
    }
    __syncthreads();
  }

  // Phase 2: softmax along j for each of the 16 rows (wave32 reductions)
  for (int row = 0; row < 16; ++row) {
    float mx = -3.4e38f;
    for (int j = lane; j < NTOK; j += 32) mx = fmaxf(mx, S[row][j]);
    mx = wave_max(mx);
    float sum = 0.f;
    for (int j = lane; j < NTOK; j += 32) {
      const float ev = __expf(S[row][j] - mx);
      S[row][j] = ev;
      sum += ev;
    }
    const float inv = 1.f / wave_sum(sum);
    for (int j = lane; j < NTOK; j += 32) {
      const float a = S[row][j] * inv;
      S[row][j] = a;
      attn[(long long)h * NN + (long long)(i0 + row) * NTOK + j] = a;
    }
  }
  __syncthreads();

  // Phase 3: node part of output: O(16x64) = attn(16x384) @ V(384x64)
  v8f oacc[4] = {{}, {}, {}, {}};
  for (int k0 = 0; k0 < NTOK; k0 += 32) {
    // stage V rows k0..k0+31 (coalesced)
#pragma unroll
    for (int t = 0; t < 16; ++t) {
      const int idx = t * 32 + lane;            // 0..511 float4 slots
      const int row = idx >> 4, c4 = (idx & 15) * 4;
      const float4 v = *(const float4*)(kv + (long long)(k0 + row) * (2 * INNERD) +
                                        INNERD + h * DHEAD + c4);
      KVs[row][c4 + 0] = v.x; KVs[row][c4 + 1] = v.y;
      KVs[row][c4 + 2] = v.z; KVs[row][c4 + 3] = v.w;
    }
    __syncthreads();
    v16bf af;                                   // attn fragment feeds 4 WMMAs
#pragma unroll
    for (int e = 0; e < 8; ++e) {
      af[e]     = f2bf(S[r][k0 + hi * 8 + e]);
      af[e + 8] = f2bf(S[r][k0 + 16 + hi * 8 + e]);
    }
#pragma unroll
    for (int nt = 0; nt < 4; ++nt) {
      v16bf bf;
#pragma unroll
      for (int e = 0; e < 16; ++e) bf[e] = f2bf(KVs[hi * 8 + e + (e >= 8 ? 8 : 0)][nt * 16 + r]);
      oacc[nt] = __builtin_amdgcn_wmma_f32_16x16x32_bf16(false, af, false, bf,
                                                         (short)0, oacc[nt], false, false);
    }
    __syncthreads();
  }
#pragma unroll
  for (int nt = 0; nt < 4; ++nt)
#pragma unroll
    for (int g = 0; g < 8; ++g)
      aout[(long long)(i0 + hi * 8 + g) * INNERD + h * DHEAD + nt * 16 + r] = oacc[nt][g];
}

// --------------------------- gated residual --------------------------------
__global__ __launch_bounds__(32)
void gated_residual_kernel(const float* __restrict__ x, const float* __restrict__ res,
                           const float* __restrict__ Wg, float* __restrict__ out)
{
  const int n = blockIdx.x, lane = threadIdx.x;
  float s = 0.f;
  for (int d = lane; d < DIMN; d += 32) {
    const float xv = x[(long long)n * DIMN + d];
    const float rv = res[(long long)n * DIMN + d];
    s += xv * Wg[d] + rv * Wg[DIMN + d] + (xv - rv) * Wg[2 * DIMN + d];
  }
  const float gate = 1.f / (1.f + __expf(-wave_sum(s)));
  for (int d = lane; d < DIMN; d += 32) {
    const float xv = x[(long long)n * DIMN + d];
    const float rv = res[(long long)n * DIMN + d];
    out[(long long)n * DIMN + d] = xv * gate + rv * (1.f - gate);
  }
}

// ---------------------------------------------------------------------------
// Host launcher
// ---------------------------------------------------------------------------
static inline void gemm(hipStream_t st,
                        const float* A, int am, int ak, long long ab,
                        const float* W, int wk, int wn, long long wb,
                        const float* bias, int bn, long long bb,
                        float* C, int csm, int csn, long long cb,
                        int M, int Nc, int K, int batch, int act, int accum)
{
  dim3 grid((unsigned)((Nc + 63) / 64), (unsigned)((M + 15) / 16), (unsigned)batch);
  gemm_wmma_bf16<<<grid, dim3(32), 0, st>>>(A, am, ak, ab, W, wk, wn, wb,
                                            bias, bn, bb, C, csm, csn, cb,
                                            M, Nc, K, act, accum);
}

extern "C" void kernel_launch(void* const* d_in, const int* in_sizes, int n_in,
                              void* d_out, int out_size, void* d_ws, size_t ws_size,
                              hipStream_t stream)
{
  (void)in_sizes; (void)n_in; (void)out_size; (void)ws_size;

  // Inputs (setup_inputs dict order; params flattened as jax pytree leaves =
  // dict keys sorted): nodes, edges, edge_ln_b, edge_ln_g, then per layer:
  // W_ekv,W_exp,W_ff1,W_ff2,W_gate_attn,W_gate_ff,W_kv,W_o,W_q,
  // attn_ln_b,attn_ln_g,b_ekv,b_exp,b_ff1,b_ff2,b_kv,b_o,b_q,ff_ln_b,ff_ln_g
  const float* nodes_in  = (const float*)d_in[0];
  const float* edges_in  = (const float*)d_in[1];
  const float* edge_ln_b = (const float*)d_in[2];
  const float* edge_ln_g = (const float*)d_in[3];

  float* nodes_out = (float*)d_out;                           // (N, DIM)
  float* edges_out = (float*)d_out + (long long)NTOK * DIMN;  // (EDIM, N, N)

  // workspace carve-up (floats)
  float* w   = (float*)d_ws;
  float* ebn = w; w += (long long)NN * EDIM;     // edges_n in (i,j,c)
  float* xln = w; w += (long long)NTOK * DIMN;   // LN(nodes)
  float* xex = w; w += (long long)NTOK * INNERD; // node expansion
  float* qb  = w; w += (long long)NTOK * INNERD; // q (roped in place)
  float* kvb = w; w += (long long)NTOK * 2 * INNERD; // [k | v]
  float* qW  = w; w += (long long)HEADS * NTOK * EDIM;  // W_ekv_h^T q_i
  float* SEb = w; w += (long long)HEADS * NN;    // edge logit term
  float* atn = w; w += (long long)HEADS * NN;    // attention weights
  float* AE  = w; w += (long long)HEADS * NTOK * EDIM;  // attn @ edges_n
  float* aou = w; w += (long long)NTOK * INNERD; // attention output (pre W_o)
  float* prj = w; w += (long long)NTOK * DIMN;   // attn out projected
  float* ffh = w; w += (long long)NTOK * FFD;    // ff hidden
  float* ffo = w; w += (long long)NTOK * DIMN;   // ff out

  // ---- edges: LayerNorm once; produces both output layout and (i,j,c) copy
  edge_ln_kernel<<<dim3((NN + 255) / 256), dim3(256), 0, stream>>>(
      edges_in, edge_ln_g, edge_ln_b, edges_out, ebn);

  for (int l = 0; l < 2; ++l) {
    const float* const* P = (const float* const*)(d_in + 4 + 20 * l);
    const float* W_ekv = (const float*)P[0];
    const float* W_exp = (const float*)P[1];
    const float* W_ff1 = (const float*)P[2];
    const float* W_ff2 = (const float*)P[3];
    const float* W_ga  = (const float*)P[4];
    const float* W_gf  = (const float*)P[5];
    const float* W_kv  = (const float*)P[6];
    const float* W_o   = (const float*)P[7];
    const float* W_q   = (const float*)P[8];
    const float* aln_b = (const float*)P[9];
    const float* aln_g = (const float*)P[10];
    const float* b_ekv = (const float*)P[11];
    const float* b_exp = (const float*)P[12];
    const float* b_ff1 = (const float*)P[13];
    const float* b_ff2 = (const float*)P[14];
    const float* b_kv  = (const float*)P[15];
    const float* b_o   = (const float*)P[16];
    const float* b_q   = (const float*)P[17];
    const float* fln_b = (const float*)P[18];
    const float* fln_g = (const float*)P[19];

    const float* cur = (l == 0) ? nodes_in : nodes_out;

    // ---- attention branch
    ln_kernel<<<dim3(NTOK), dim3(32), 0, stream>>>(cur, aln_g, aln_b, xln, DIMN);

    // x = LN(nodes) @ W_exp + b_exp                        (384x256)x(256x512)
    gemm(stream, xln, DIMN, 1, 0, W_exp, INNERD, 1, 0, b_exp, 1, 0,
         xex, INNERD, 1, 0, NTOK, INNERD, DIMN, 1, 0, 0);
    // q = x @ W_q + b_q                                    (384x512)x(512x512)
    gemm(stream, xex, INNERD, 1, 0, W_q, INNERD, 1, 0, b_q, 1, 0,
         qb, INNERD, 1, 0, NTOK, INNERD, INNERD, 1, 0, 0);
    // kv = x @ W_kv + b_kv                                 (384x512)x(512x1024)
    gemm(stream, xex, INNERD, 1, 0, W_kv, 2 * INNERD, 1, 0, b_kv, 1, 0,
         kvb, 2 * INNERD, 1, 0, NTOK, 2 * INNERD, INNERD, 1, 0, 0);
    // rope(q), rope(k) in place
    rope_kernel<<<dim3((NTOK * INNERD / 2 + 255) / 256), dim3(256), 0, stream>>>(qb, kvb);

    // qW[h,i,c] = sum_d W_ekv[c, h*64+d] * q[h,i,d]   (batched over heads)
    gemm(stream, qb, INNERD, 1, /*ab=*/DHEAD,
         W_ekv, /*wk=*/1, /*wn=*/INNERD, /*wb=*/DHEAD,
         nullptr, 0, 0,
         qW, EDIM, 1, (long long)NTOK * EDIM,
         NTOK, EDIM, DHEAD, HEADS, 0, 0);

    // SE[h,i,j] = edges_n[i,j,:] . qW[h,i,:]          (batched over i rows)
    gemm(stream, ebn, EDIM, 1, (long long)NTOK * EDIM,      // A[j,c] per i
         qW, 1, (long long)NTOK * EDIM, EDIM,               // W[c,h] per i
         nullptr, 0, 0,
         SEb, 1, NN, NTOK,                                  // C[j,h] -> SE[h,i,j]
         NTOK, HEADS, EDIM, NTOK, 0, 0);

    // fused logits + softmax + attn@V (TDM-staged q tile)
    attn_kernel<<<dim3(NTOK / 16, HEADS), dim3(32), 0, stream>>>(qb, kvb, SEb, atn, aou);

    // AE[h,i,c] = sum_j attn[h,i,j] * edges_n[i,j,c]  (batched over i rows)
    gemm(stream, atn, NN, 1, NTOK,                          // A[h,j] per i
         ebn, EDIM, 1, (long long)NTOK * EDIM,              // W[j,c] per i
         nullptr, 0, 0,
         AE, (long long)NTOK * EDIM, 1, EDIM,               // C[h,c] -> AE[h,i,c]
         HEADS, EDIM, NTOK, NTOK, 0, 0);

    // aout[i, h*64+d] += AE[h,i,:] @ W_ekv[:, h*64+d] + b_ekv[h*64+d]
    gemm(stream, AE, EDIM, 1, (long long)NTOK * EDIM,
         W_ekv, INNERD, 1, DHEAD,
         b_ekv, 1, DHEAD,
         aou, INNERD, 1, DHEAD,
         NTOK, DHEAD, EDIM, HEADS, 0, /*accum=*/1);

    // proj = aout @ W_o + b_o
    gemm(stream, aou, INNERD, 1, 0, W_o, DIMN, 1, 0, b_o, 1, 0,
         prj, DIMN, 1, 0, NTOK, DIMN, INNERD, 1, 0, 0);

    gated_residual_kernel<<<dim3(NTOK), dim3(32), 0, stream>>>(prj, cur, W_ga, nodes_out);

    // ---- feedforward branch
    ln_kernel<<<dim3(NTOK), dim3(32), 0, stream>>>(nodes_out, fln_g, fln_b, xln, DIMN);
    gemm(stream, xln, DIMN, 1, 0, W_ff1, FFD, 1, 0, b_ff1, 1, 0,
         ffh, FFD, 1, 0, NTOK, FFD, DIMN, 1, /*gelu*/1, 0);
    gemm(stream, ffh, FFD, 1, 0, W_ff2, DIMN, 1, 0, b_ff2, 1, 0,
         ffo, DIMN, 1, 0, NTOK, DIMN, FFD, 1, 0, 0);
    gated_residual_kernel<<<dim3(NTOK), dim3(32), 0, stream>>>(ffo, nodes_out, W_gf, nodes_out);
  }
}
